// DifferentialGQA_73151882985489
// MI455X (gfx1250) — matrix operand order repaired and checked by reading, not compiled, not run on hardware
//
#include <hip/hip_runtime.h>
#include <hip/hip_bf16.h>
#include <math.h>

#define HQ     32
#define HKV    8
#define DH     64
#define GROUPS 4
#define NPAIR  16
#define CAPF   50.0f
#define HIDW   2048

typedef __bf16 bf16;
typedef __attribute__((ext_vector_type(16))) __bf16 v16bf;
typedef __attribute__((ext_vector_type(8)))  __bf16 v8bf;
typedef __attribute__((ext_vector_type(8)))  float  v8f;

#define WMMA_BF16(a, b, c) \
  __builtin_amdgcn_wmma_f32_16x16x32_bf16(false, (a), false, (b), (short)0, (c), false, false)

__device__ __forceinline__ v8f zero_v8f() {
  v8f z;
#pragma unroll
  for (int i = 0; i < 8; ++i) z[i] = 0.0f;
  return z;
}

// A fragment: 16x32 bf16 tile, row-major source [16, >=32].
// ISA layout kb = (v>>2)*16 + half*8 + (v&3)*2 == two contiguous 16B runs:
//   elements [m*ld + half*8, +8) and [m*ld + 16 + half*8, +8)
__device__ __forceinline__ v16bf frag_a_rm(const bf16* p, int ld, int lane) {
  int m = lane & 15, half = lane >> 4;
  const bf16* base = p + m * ld + half * 8;
  v8bf lo = *(const v8bf*)(base);
  v8bf hi = *(const v8bf*)(base + 16);
  return __builtin_shufflevector(lo, hi, 0, 1, 2, 3, 4, 5, 6, 7,
                                 8, 9, 10, 11, 12, 13, 14, 15);
}

// B fragment: 32x16 tile, source stored row-major as B^T = [N, K].
// Layout matches frag_a_rm: two contiguous 16B runs at n*ld + half*8 and n*ld + 16 + half*8.
__device__ __forceinline__ v16bf frag_b_nk(const bf16* p, int ld, int lane) {
  int n = lane & 15, half = lane >> 4;
  const bf16* base = p + n * ld + half * 8;
  v8bf lo = *(const v8bf*)(base);
  v8bf hi = *(const v8bf*)(base + 16);
  return __builtin_shufflevector(lo, hi, 0, 1, 2, 3, 4, 5, 6, 7,
                                 8, 9, 10, 11, 12, 13, 14, 15);
}

// ---------------------------------------------------------------------------
// C(MxN f32) = A(MxK bf16 row-major) * Bt(NxK bf16 row-major)^T
// 128 threads = 4 waves; each wave: 32x64 tile (2x4 WMMA blocking);
// block covers 64 rows x 128 cols.
// ---------------------------------------------------------------------------
__global__ __launch_bounds__(128) void gemm_bf16_nt(
    const bf16* __restrict__ A, const bf16* __restrict__ Bt,
    float* __restrict__ C, int M, int N, int K)
{
  int lane = threadIdx.x & 31;
  int wave = threadIdx.x >> 5;
  int m0 = blockIdx.y * 64 + (wave >> 1) * 32;
  int n0 = blockIdx.x * 128 + (wave & 1) * 64;
  if (m0 >= M || n0 >= N) return;
  v8f acc[2][4];
#pragma unroll
  for (int i = 0; i < 2; ++i)
#pragma unroll
    for (int j = 0; j < 4; ++j) acc[i][j] = zero_v8f();

  for (int k0 = 0; k0 < K; k0 += 32) {
    v16bf a0 = frag_a_rm(A + (size_t)m0 * K + k0, K, lane);
    v16bf a1 = frag_a_rm(A + (size_t)(m0 + 16) * K + k0, K, lane);
    v16bf b[4];
#pragma unroll
    for (int j = 0; j < 4; ++j)
      b[j] = frag_b_nk(Bt + (size_t)(n0 + j * 16) * K + k0, K, lane);
#pragma unroll
    for (int j = 0; j < 4; ++j) {
      acc[0][j] = WMMA_BF16(a0, b[j], acc[0][j]);
      acc[1][j] = WMMA_BF16(a1, b[j], acc[1][j]);
    }
  }
  int n = lane & 15, half = lane >> 4;
#pragma unroll
  for (int i = 0; i < 2; ++i)
#pragma unroll
    for (int j = 0; j < 4; ++j)
#pragma unroll
      for (int r = 0; r < 8; ++r)
        C[(size_t)(m0 + i * 16 + r + 8 * half) * N + n0 + j * 16 + n] = acc[i][j][r];
}

// ---------------------------------------------------------------------------
// Elementwise helpers
// ---------------------------------------------------------------------------
__global__ void f32_to_bf16_k(const float* __restrict__ in, bf16* __restrict__ out, int n) {
  int i = blockIdx.x * blockDim.x + threadIdx.x;
  if (i < n) out[i] = (bf16)in[i];
}

// in [K, N] f32 -> out [N, K] bf16
__global__ void transpose_to_bf16_k(const float* __restrict__ in, bf16* __restrict__ out,
                                    int K, int N) {
  int i = blockIdx.x * blockDim.x + threadIdx.x;
  if (i >= K * N) return;
  int k = i / N, n = i - k * N;
  out[(size_t)n * K + k] = (bf16)in[i];
}

// RoPE on [L, NH*64] f32 -> bf16 ; cos/sin: [L, 32] f32
__global__ void rope_to_bf16_k(const float* __restrict__ in, const float* __restrict__ cosb,
                               const float* __restrict__ sinb, bf16* __restrict__ out,
                               int L, int NH) {
  int idx = blockIdx.x * blockDim.x + threadIdx.x;
  int total = L * NH * (DH / 2);
  if (idx >= total) return;
  int d = idx & 31;
  int h = (idx >> 5) % NH;
  int l = idx / (NH * (DH / 2));
  const float* row = in + (size_t)l * NH * DH + h * DH;
  bf16* orow = out + (size_t)l * NH * DH + h * DH;
  float c = cosb[l * (DH / 2) + d];
  float s = sinb[l * (DH / 2) + d];
  float x1 = row[d], x2 = row[d + DH / 2];
  orow[d]          = (bf16)(x1 * c - x2 * s);
  orow[d + DH / 2] = (bf16)(x1 * s + x2 * c);
}

__global__ void zero_f32_k(float* p, int n) {
  int i = blockIdx.x * blockDim.x + threadIdx.x;
  if (i < n) p[i] = 0.0f;
}

// ---------------------------------------------------------------------------
// Lambda scalar: four dot products reduced over (L, pair-heads, D)
// ---------------------------------------------------------------------------
__global__ __launch_bounds__(256) void lambda_reduce_k(
    const bf16* __restrict__ qbf, const bf16* __restrict__ kbf,
    const float* __restrict__ lq1, const float* __restrict__ lk1,
    const float* __restrict__ lq2, const float* __restrict__ lk2,
    float* __restrict__ part, int L)
{
  int idx = blockIdx.x * blockDim.x + threadIdx.x;
  float a0 = 0.f, a1 = 0.f, a2 = 0.f, a3 = 0.f;
  int total = L * NPAIR * DH;
  if (idx < total) {
    int d  = idx & (DH - 1);
    int ph = (idx / DH) % NPAIR;
    int l  = idx / (DH * NPAIR);
    int h1 = 2 * ph, h2 = 2 * ph + 1;
    int kv1 = h1 / GROUPS, kv2 = h2 / GROUPS;
    a0 = (float)qbf[(size_t)l * HQ * DH + h1 * DH + d] * lq1[d];
    a1 = (float)kbf[(size_t)l * HKV * DH + kv1 * DH + d] * lk1[d];
    a2 = (float)qbf[(size_t)l * HQ * DH + h2 * DH + d] * lq2[d];
    a3 = (float)kbf[(size_t)l * HKV * DH + kv2 * DH + d] * lk2[d];
  }
  __shared__ float red[4][256];
  int t = threadIdx.x;
  red[0][t] = a0; red[1][t] = a1; red[2][t] = a2; red[3][t] = a3;
  __syncthreads();
  for (int s = 128; s > 0; s >>= 1) {
    if (t < s) {
#pragma unroll
      for (int j = 0; j < 4; ++j) red[j][t] += red[j][t + s];
    }
    __syncthreads();
  }
  if (t == 0) {
#pragma unroll
    for (int j = 0; j < 4; ++j) atomicAdd(&part[j], red[j][0]);
  }
}

__global__ void lambda_finalize_k(float* part, int L) {
  float inv = 1.0f / (float)L;   // mean over (B=1, L); heads/dims are summed
  float d1 = fminf(fmaxf(part[0] * inv, -10.f), 10.f);
  float d2 = fminf(fmaxf(part[1] * inv, -10.f), 10.f);
  float d3 = fminf(fmaxf(part[2] * inv, -10.f), 10.f);
  float d4 = fminf(fmaxf(part[3] * inv, -10.f), 10.f);
  float lam_init = 0.8f - 0.6f * expf(-1.2f);  // layer_idx = 4
  float lam = expf(d1) * expf(d2) - expf(d3) * expf(d4) + lam_init;
  part[4] = fminf(fmaxf(lam, 0.0f), 1.0f);
}

// ---------------------------------------------------------------------------
// Differential attention. One wave per (pair-head, 16-row Q block).
// With GROUPS=4 and h1=2*ph even: kv(h1)==kv(h2), so K1==K2, V1==V2 exactly;
// K fragments are shared by s1/s2 and P·V is computed once (out1==out2).
// Softcap bounds scores to +-50 so exp() without max-subtraction is safe.
// Pass 1: Z1/Z2 row sums. Pass 2: p = max(e1/Z1 - lam*e2/Z2, 0), P·V.
// vT: V pre-transposed to [HKV*DH, L] so PV B-fragments load contiguously.
// out layout: [L, 2048], col = ph*128 + (0..63 out1 | 64..127 out2)
// ---------------------------------------------------------------------------
__global__ __launch_bounds__(256) void diff_attn_k(
    const bf16* __restrict__ qbf, const bf16* __restrict__ kbf,
    const bf16* __restrict__ vT, const float* __restrict__ lam_ptr,
    float* __restrict__ out, int L)
{
  __shared__ __align__(16) bf16 pstage[8][16 * 32];
  int lane = threadIdx.x & 31;
  int wave = threadIdx.x >> 5;
  int gw = blockIdx.x * 8 + wave;
  int nqb = L / 16;
  int ph = gw / nqb;
  int q0 = (gw - ph * nqb) * 16;
  int h1 = 2 * ph, h2 = h1 + 1;
  int kv = h1 / GROUPS;                // == h2 / GROUPS for GROUPS=4
  const float scale = 0.125f;          // 1/sqrt(64)
  const float icap = 1.0f / CAPF;
  const float lam = *lam_ptr;
  const int ldq = HQ * DH, ldkv = HKV * DH;

  v16bf qa1[2], qa2[2];
#pragma unroll
  for (int kk = 0; kk < 2; ++kk) {
    qa1[kk] = frag_a_rm(qbf + (size_t)q0 * ldq + h1 * DH + kk * 32, ldq, lane);
    qa2[kk] = frag_a_rm(qbf + (size_t)q0 * ldq + h2 * DH + kk * 32, ldq, lane);
  }

  int n = lane & 15, half = lane >> 4;
  float z1[8], z2[8];
#pragma unroll
  for (int r = 0; r < 8; ++r) { z1[r] = 0.f; z2[r] = 0.f; }

  int kend = q0 + 16;  // causal: keys >= q0+16 never contribute

  // ---------------- pass 1: per-lane column partials of exp sums ------------
  for (int kc = 0; kc < kend; kc += 16) {
    v8f s1 = zero_v8f(), s2 = zero_v8f();
    v16bf b0 = frag_b_nk(kbf + (size_t)kc * ldkv + kv * DH + 0,  ldkv, lane);
    v16bf b1 = frag_b_nk(kbf + (size_t)kc * ldkv + kv * DH + 32, ldkv, lane);
    s1 = WMMA_BF16(qa1[0], b0, s1);
    s1 = WMMA_BF16(qa1[1], b1, s1);
    s2 = WMMA_BF16(qa2[0], b0, s2);
    s2 = WMMA_BF16(qa2[1], b1, s2);
#pragma unroll
    for (int r = 0; r < 8; ++r) {
      int i = q0 + r + 8 * half;
      int j = kc + n;
      float x1 = CAPF * tanhf(s1[r] * scale * icap);
      float x2 = CAPF * tanhf(s2[r] * scale * icap);
      z1[r] += (j <= i) ? __expf(x1) : 0.0f;
      z2[r] += (j <= i) ? __expf(x2) : 0.0f;
    }
  }
  // one butterfly reduction over the 16 lanes of each half
#pragma unroll
  for (int r = 0; r < 8; ++r) {
#pragma unroll
    for (int m = 8; m >= 1; m >>= 1) {
      z1[r] += __shfl_xor(z1[r], m, 16);
      z2[r] += __shfl_xor(z2[r], m, 16);
    }
  }

  // ---------------- pass 2: differential probs + P·V ----------------
  v8f o[4];
#pragma unroll
  for (int dt = 0; dt < 4; ++dt) o[dt] = zero_v8f();

  int kend2 = (kend + 31) & ~31;
  for (int kc = 0; kc < kend2; kc += 32) {
#pragma unroll
    for (int sub = 0; sub < 2; ++sub) {
      int kcs = kc + sub * 16;
      v8f s1 = zero_v8f(), s2 = zero_v8f();
      v16bf b0 = frag_b_nk(kbf + (size_t)kcs * ldkv + kv * DH + 0,  ldkv, lane);
      v16bf b1 = frag_b_nk(kbf + (size_t)kcs * ldkv + kv * DH + 32, ldkv, lane);
      s1 = WMMA_BF16(qa1[0], b0, s1);
      s1 = WMMA_BF16(qa1[1], b1, s1);
      s2 = WMMA_BF16(qa2[0], b0, s2);
      s2 = WMMA_BF16(qa2[1], b1, s2);
#pragma unroll
      for (int r = 0; r < 8; ++r) {
        int i = q0 + r + 8 * half;
        int j = kcs + n;
        float x1 = CAPF * tanhf(s1[r] * scale * icap);
        float x2 = CAPF * tanhf(s2[r] * scale * icap);
        float e1 = (j <= i) ? __expf(x1) : 0.0f;
        float e2 = (j <= i) ? __expf(x2) : 0.0f;
        float p = fmaxf(e1 / z1[r] - lam * e2 / z2[r], 0.0f);
        pstage[wave][(r + 8 * half) * 32 + sub * 16 + n] = (bf16)p;
      }
    }
    __builtin_amdgcn_wave_barrier();     // wave-local LDS transpose staging
    v16bf pa = frag_a_rm(&pstage[wave][0], 32, lane);
    __builtin_amdgcn_wave_barrier();
#pragma unroll
    for (int dt = 0; dt < 4; ++dt) {
      v16bf vb = frag_b_nk(vT + (size_t)(kv * DH + dt * 16) * L + kc, L, lane);
      o[dt] = WMMA_BF16(pa, vb, o[dt]);
    }
  }

#pragma unroll
  for (int dt = 0; dt < 4; ++dt) {
#pragma unroll
    for (int r = 0; r < 8; ++r) {
      int row = q0 + r + 8 * half;
      float* orow = out + (size_t)row * (NPAIR * 2 * DH) + ph * (2 * DH);
      orow[dt * 16 + n]      = o[dt][r];   // out1
      orow[DH + dt * 16 + n] = o[dt][r];   // out2 (v1==v2 -> identical)
    }
  }
}

// ---------------------------------------------------------------------------
// RMS-norm over each 128-wide pair group, scale by subln weight & (1-lam_init)
// ---------------------------------------------------------------------------
__global__ void rms_epilogue_k(const float* __restrict__ in, const float* __restrict__ w,
                               bf16* __restrict__ out, int L) {
  int g = blockIdx.x * blockDim.x + threadIdx.x;
  if (g >= L * NPAIR) return;
  const float* p = in + (size_t)g * (2 * DH);
  float ss = 0.f;
#pragma unroll 8
  for (int c = 0; c < 2 * DH; ++c) { float v = p[c]; ss += v * v; }
  float rinv = rsqrtf(ss * (1.0f / (2 * DH)) + 1e-6f);
  float post = 1.0f - (0.8f - 0.6f * expf(-1.2f));
  bf16* o = out + (size_t)g * (2 * DH);
#pragma unroll 8
  for (int c = 0; c < 2 * DH; ++c) o[c] = (bf16)(p[c] * rinv * w[c] * post);
}

// ---------------------------------------------------------------------------
extern "C" void kernel_launch(void* const* d_in, const int* in_sizes, int n_in,
                              void* d_out, int out_size, void* d_ws, size_t ws_size,
                              hipStream_t stream) {
  const float* x    = (const float*)d_in[0];
  const float* cosb = (const float*)d_in[1];
  const float* sinb = (const float*)d_in[2];
  // d_in[3] = mask (bool) — causal mask computed analytically, unused
  const float* Wq   = (const float*)d_in[4];
  const float* Wk   = (const float*)d_in[5];
  const float* Wv   = (const float*)d_in[6];
  const float* Wo   = (const float*)d_in[7];
  const float* lq1  = (const float*)d_in[8];
  const float* lk1  = (const float*)d_in[9];
  const float* lq2  = (const float*)d_in[10];
  const float* lk2  = (const float*)d_in[11];
  const float* subw = (const float*)d_in[12];
  float* outp = (float*)d_out;

  const int L = 2048;
  const int KVW = HKV * DH;  // 512

  char* w = (char*)d_ws;
  auto alloc = [&](size_t bytes) {
    char* p = w;
    w += (bytes + 255) & ~(size_t)255;
    return p;
  };
  bf16*  xbf  = (bf16*)alloc((size_t)L * HIDW * 2);
  bf16*  Wqt  = (bf16*)alloc((size_t)HIDW * HIDW * 2);
  bf16*  Wkt  = (bf16*)alloc((size_t)KVW * HIDW * 2);
  bf16*  Wvt  = (bf16*)alloc((size_t)KVW * HIDW * 2);
  bf16*  Wot  = (bf16*)alloc((size_t)HIDW * HIDW * 2);
  float* qf32 = (float*)alloc((size_t)L * HIDW * 4);
  float* kf32 = (float*)alloc((size_t)L * KVW * 4);
  float* vf32 = (float*)alloc((size_t)L * KVW * 4);
  bf16*  qbf  = (bf16*)alloc((size_t)L * HIDW * 2);
  bf16*  kbf  = (bf16*)alloc((size_t)L * KVW * 2);
  bf16*  vbT  = (bf16*)alloc((size_t)KVW * L * 2);   // V^T: [512, L]
  float* part = (float*)alloc(256);
  // reuse: qf32 free after RoPE -> attention output; kf32+vf32 free -> ao bf16
  float* aof  = qf32;                 // [L, 2048] f32
  bf16*  aobf = (bf16*)kf32;          // [L, 2048] bf16 (8 MB, spans kf32+vf32)

  const int TB = 256;
  // 1) conversions / transposes
  {
    int n = L * HIDW;
    f32_to_bf16_k<<<(n + TB - 1) / TB, TB, 0, stream>>>(x, xbf, n);
    int nw = HIDW * HIDW;
    transpose_to_bf16_k<<<(nw + TB - 1) / TB, TB, 0, stream>>>(Wq, Wqt, HIDW, HIDW);
    int nk = HIDW * KVW;
    transpose_to_bf16_k<<<(nk + TB - 1) / TB, TB, 0, stream>>>(Wk, Wkt, HIDW, KVW);
    transpose_to_bf16_k<<<(nk + TB - 1) / TB, TB, 0, stream>>>(Wv, Wvt, HIDW, KVW);
    transpose_to_bf16_k<<<(nw + TB - 1) / TB, TB, 0, stream>>>(Wo, Wot, HIDW, HIDW);
  }
  // 2) Q/K/V projections
  gemm_bf16_nt<<<dim3(HIDW / 128, L / 64), 128, 0, stream>>>(xbf, Wqt, qf32, L, HIDW, HIDW);
  gemm_bf16_nt<<<dim3(KVW / 128, L / 64), 128, 0, stream>>>(xbf, Wkt, kf32, L, KVW, HIDW);
  gemm_bf16_nt<<<dim3(KVW / 128, L / 64), 128, 0, stream>>>(xbf, Wvt, vf32, L, KVW, HIDW);
  // 3) RoPE(q,k) -> bf16 ; V -> bf16 transposed to [512, L]
  {
    int nq = L * HQ * (DH / 2);
    rope_to_bf16_k<<<(nq + TB - 1) / TB, TB, 0, stream>>>(qf32, cosb, sinb, qbf, L, HQ);
    int nk = L * HKV * (DH / 2);
    rope_to_bf16_k<<<(nk + TB - 1) / TB, TB, 0, stream>>>(kf32, cosb, sinb, kbf, L, HKV);
    int nv = L * KVW;
    transpose_to_bf16_k<<<(nv + TB - 1) / TB, TB, 0, stream>>>(vf32, vbT, L, KVW);
  }
  // 4) lambda scalar
  zero_f32_k<<<1, 32, 0, stream>>>(part, 8);
  {
    int n = L * NPAIR * DH;
    lambda_reduce_k<<<(n + TB - 1) / TB, TB, 0, stream>>>(qbf, kbf, lq1, lk1, lq2, lk2, part, L);
  }
  lambda_finalize_k<<<1, 1, 0, stream>>>(part, L);
  // 5) differential attention (2048 waves, 8 waves/block)
  diff_attn_k<<<(NPAIR * (L / 16)) / 8, 256, 0, stream>>>(qbf, kbf, vbT, part + 4, aof, L);
  // 6) RMS-norm epilogue
  {
    int n = L * NPAIR;
    rms_epilogue_k<<<(n + TB - 1) / TB, TB, 0, stream>>>(aof, subw, aobf, L);
  }
  // 7) output projection
  gemm_bf16_nt<<<dim3(HIDW / 128, L / 64), 128, 0, stream>>>(aobf, Wot, outp, L, HIDW, HIDW);
}
